// CausalMHA_RoPE_GQA_30709016166848
// MI455X (gfx1250) — matrix operand order repaired and checked by reading, not compiled, not run on hardware
//
#include <hip/hip_runtime.h>
#include <hip/hip_bf16.h>

// ---------------------------------------------------------------------------
// CausalMHA + RoPE + GQA for MI455X (gfx1250, wave32, WMMA 16x16x32 f16,
// double-buffered async global->LDS staging: GLOBAL_LOAD_ASYNC_TO_LDS_B128
// overlapped with WMMA compute, one barrier per pipeline stage).
//
// Pipeline (all on `stream`):
//   1) cast x, Wq, Wk, Wv, Wo   fp32 -> fp16 (workspace)
//   2) Q = x@Wq^T, K = x@Wk^T, V = x@Wv^T      (WMMA NT GEMM, f16 out)
//   3) RoPE(Q), RoPE(K)
//   4) flash attention (causal, GQA kv = h>>1)  -> Y f16 [B,T,NH,HD]
//   5) out = Y @ Wo^T                           (WMMA NT GEMM, f32 out)
// ---------------------------------------------------------------------------

typedef _Float16 half_t;
typedef __attribute__((ext_vector_type(16))) _Float16 v16h;
typedef __attribute__((ext_vector_type(8)))  _Float16 v8h;
typedef __attribute__((ext_vector_type(8)))  float    v8f;

constexpr int D_MODEL = 1024;
constexpr int N_HEAD  = 16;
constexpr int N_KV    = 8;
constexpr int HD      = 64;      // head dim
constexpr int B_SZ    = 2;
constexpr int T_SZ    = 2048;
constexpr int BT      = B_SZ * T_SZ;

// ---- async global->LDS (ASYNCcnt) -----------------------------------------
__device__ inline void async_copy_b128(unsigned lds_off, const void* gaddr) {
  asm volatile("global_load_async_to_lds_b128 %0, %1, off"
               :: "v"(lds_off), "v"(gaddr) : "memory");
}
__device__ inline void wait_async0() {
  asm volatile("s_wait_asynccnt 0x0" ::: "memory");
}
// generic LDS pointer -> LDS byte offset (low 32 bits of the address)
__device__ inline unsigned lds_off_of(const void* p) {
  return (unsigned)(unsigned long long)p;
}

// ---- fragment helpers (layouts per CDNA5 ISA 7.12.2, wave32) --------------
// A (16x32 f16): lane r / r+16 holds row r; VGPR0-3 = K {g*8..g*8+7},
// VGPR4-7 = K {16+g*8..16+g*8+7}, g = lane>>4. Two 16B loads per lane.
__device__ inline v16h load_a_pair(const half_t* p /* row base + g*8 */) {
  union { v16h v; v8h h[2]; } u;
  u.h[0] = *(const v8h*)(p);
  u.h[1] = *(const v8h*)(p + 16);
  return u.v;
}
// B (32x16 f16) NT: lane n+16g holds K {g*8..g*8+7} and {16+g*8..16+g*8+7}
__device__ inline v16h load_b_pair(const half_t* p /* row base + g*8 */) {
  union { v16h v; v8h h[2]; } u;
  u.h[0] = *(const v8h*)(p);
  u.h[1] = *(const v8h*)(p + 16);
  return u.v;
}

// ---- 1) fp32 -> fp16 ------------------------------------------------------
__global__ void cast_f32_to_f16(const float* __restrict__ src, half_t* __restrict__ dst, int n) {
  int i = blockIdx.x * blockDim.x + threadIdx.x;
  if (i < n) dst[i] = (half_t)src[i];
}

// ---- 2/5) NT WMMA GEMM: C[M,N] = A[M,K] * W[N,K]^T ------------------------
// block = 256 threads = 8 waves (4 along M x 2 along N); wave tile 32x64,
// block tile 128x128. 64-wide k-slabs double-buffered in LDS via async
// copies overlapped with the 16 WMMAs of the previous slab.
template <bool F32OUT>
__global__ __launch_bounds__(256) void gemm_nt_wmma(
    const half_t* __restrict__ A, const half_t* __restrict__ W,
    void* __restrict__ Cout, int M, int N, int K) {
  __shared__ half_t As[2][128][72];   // 128 rows x 64 halves, pad 8
  __shared__ half_t Bs[2][128][72];

  const int tid  = threadIdx.x;
  const int lane = tid & 31;
  const int wid  = tid >> 5;
  const int wm   = wid & 3;          // 0..3 (M)
  const int wn   = wid >> 2;         // 0..1 (N)
  const int m0   = blockIdx.x * 128;
  const int n0   = blockIdx.y * 128;
  const int r = lane & 15, g = lane >> 4;

  // staging: thread -> (row, 64B half of the 128B k-slab row), 4 x b128 each
  const int srow = tid & 127;
  const int scb  = (tid >> 7) * 64;                 // 0 or 64 bytes

  auto stage = [&](int buf, int k0) {
    const char* ga = (const char*)(A + (size_t)(m0 + srow) * K + k0) + scb;
    const char* gb = (const char*)(W + (size_t)(n0 + srow) * K + k0) + scb;
    unsigned la = lds_off_of(&As[buf][srow][0]) + scb;
    unsigned lb = lds_off_of(&Bs[buf][srow][0]) + scb;
#pragma unroll
    for (int i = 0; i < 4; ++i) {
      async_copy_b128(la + i * 16, ga + i * 16);
      async_copy_b128(lb + i * 16, gb + i * 16);
    }
  };

  v8f acc[2][4] = {};
  stage(0, 0);
  for (int k0 = 0; k0 < K; k0 += 64) {
    wait_async0();        // copy issued last iteration is now needed
    __syncthreads();      // everyone's staging of `cur` complete
    const int cur = (k0 >> 6) & 1;
    if (k0 + 64 < K) {
      __builtin_prefetch((const char*)(W + (size_t)(n0 + srow) * K + k0) + 384, 0, 1);
      stage(cur ^ 1, k0 + 64);      // overlapped with compute below
    }
#pragma unroll
    for (int kk = 0; kk < 64; kk += 32) {
      v16h a0 = load_a_pair(&As[cur][wm * 32 +      r][kk + g * 8]);
      v16h a1 = load_a_pair(&As[cur][wm * 32 + 16 + r][kk + g * 8]);
#pragma unroll
      for (int nf = 0; nf < 4; ++nf) {
        v16h b = load_b_pair(&Bs[cur][wn * 64 + nf * 16 + r][kk + g * 8]);
        acc[0][nf] = __builtin_amdgcn_wmma_f32_16x16x32_f16(
            false, a0, false, b, (short)0, acc[0][nf], false, false);
        acc[1][nf] = __builtin_amdgcn_wmma_f32_16x16x32_f16(
            false, a1, false, b, (short)0, acc[1][nf], false, false);
      }
    }
  }

  // C layout: lane r/(r+16), VGPR v -> row v + g*8, col r
#pragma unroll
  for (int mf = 0; mf < 2; ++mf)
#pragma unroll
    for (int nf = 0; nf < 4; ++nf)
#pragma unroll
      for (int v = 0; v < 8; ++v) {
        size_t row = (size_t)(m0 + wm * 32 + mf * 16 + g * 8 + v);
        size_t col = (size_t)(n0 + wn * 64 + nf * 16 + r);
        float  val = acc[mf][nf][v];
        if (F32OUT) ((float*)Cout)[row * N + col]  = val;
        else        ((half_t*)Cout)[row * N + col] = (half_t)val;
      }
}

// ---- 3) RoPE on f16 [B,T,H,64], interleaved pairs -------------------------
__global__ void rope_f16(half_t* __restrict__ q, int nheads) {
  int idx = blockIdx.x * blockDim.x + threadIdx.x;   // (bt, h, i) flat
  int i   = idx & 31;                                // pair index 0..31
  int h   = (idx >> 5) % nheads;
  int bt  = idx / (32 * nheads);
  if (bt >= BT) return;
  int t = bt & (T_SZ - 1);
  float inv_freq = __powf(10000.0f, -(float)(2 * i) / (float)HD);
  float ang = (float)t * inv_freq;
  float s = __sinf(ang), c = __cosf(ang);
  half_t* p = q + ((size_t)bt * nheads + h) * HD + 2 * i;
  float x1 = (float)p[0], x2 = (float)p[1];
  p[0] = (half_t)(x1 * c - x2 * s);
  p[1] = (half_t)(x1 * s + x2 * c);
}

// ---- 4) flash attention (causal, GQA) -------------------------------------
// grid = (T/64, NH, B), block = 128 (4 waves); wave owns 16 q rows.
// K tile async-staged, V^T tile transpose-staged, both double-buffered and
// overlapped with the S / softmax / PV compute of the previous tile.
__global__ __launch_bounds__(128) void flash_attn_wmma(
    const half_t* __restrict__ Q, const half_t* __restrict__ K,
    const half_t* __restrict__ V, half_t* __restrict__ Y) {
  const int lane = threadIdx.x & 31;
  const int wid  = threadIdx.x >> 5;          // 0..3
  const int i0   = blockIdx.x * 64;           // q tile base
  const int h    = blockIdx.y;                // 0..15
  const int b    = blockIdx.z;
  const int kv   = h >> 1;                    // G = NH/NKV = 2

  const int ldq = N_HEAD * HD;                // 1024
  const int ldk = N_KV * HD;                  // 512
  const half_t* Qb = Q + (size_t)b * T_SZ * ldq + h * HD;
  const half_t* Kb = K + (size_t)b * T_SZ * ldk + kv * HD;
  const half_t* Vb = V + (size_t)b * T_SZ * ldk + kv * HD;

  __shared__ half_t Kt[2][64][72];            // [s][d], pad rows to 144B
  __shared__ half_t Vt[2][64][72];            // [d][s], pad rows to 144B
  __shared__ half_t Pt[4][16][64];            // per-wave P tile (q x s)

  const int qrow0 = i0 + wid * 16;
  const int r = lane & 15, g = lane >> 4;

  auto stage_k = [&](int buf, int j0) {       // async, tracked by ASYNCcnt
    int s  = threadIdx.x & 63;
    int cb = ((threadIdx.x >> 6) & 1) * 64;   // 0 / 64 bytes of the 128B row
    const char* gp = (const char*)(Kb + (size_t)(j0 + s) * ldk) + cb;
    unsigned lp = lds_off_of(&Kt[buf][s][0]) + cb;
#pragma unroll
    for (int i = 0; i < 4; ++i)
      async_copy_b128(lp + i * 16, gp + i * 16);
  };
  auto stage_v = [&](int buf, int j0) {       // manual transpose
    int s  = threadIdx.x & 63;
    int dh = (threadIdx.x >> 6) & 1;          // 32-wide d chunk
    const half_t* vp = Vb + (size_t)(j0 + s) * ldk + dh * 32;
#pragma unroll
    for (int d = 0; d < 32; ++d)
      Vt[buf][dh * 32 + d][s] = vp[d];
  };

  // Q fragments are loop-invariant: 16 rows x 64 d = 2 k-steps
  v16h qa[2];
  qa[0] = load_a_pair(Qb + (size_t)(qrow0 + r) * ldq +      g * 8);
  qa[1] = load_a_pair(Qb + (size_t)(qrow0 + r) * ldq + 32 + g * 8);

  float mrow[8], lrow[8];
  v8f   o[4] = {};
#pragma unroll
  for (int v = 0; v < 8; ++v) { mrow[v] = -1e30f; lrow[v] = 0.0f; }

  const int jend = i0 + 64;                   // causal bound (exclusive)
  stage_k(0, 0);
  stage_v(0, 0);
  for (int j0 = 0; j0 < jend; j0 += 64) {
    wait_async0();
    __syncthreads();                          // Kt/Vt[cur] ready for everyone
    const int cur = (j0 >> 6) & 1;
    if (j0 + 64 < jend) {                     // overlapped with compute below
      stage_k(cur ^ 1, j0 + 64);
      stage_v(cur ^ 1, j0 + 64);
    }

    // -- S = (Q @ K^T) * scale, causal mask --
    v8f sacc[4] = {};
#pragma unroll
    for (int kk = 0; kk < 2; ++kk)
#pragma unroll
      for (int nf = 0; nf < 4; ++nf) {
        v16h bk = load_b_pair(&Kt[cur][nf * 16 + r][kk * 32 + g * 8]);
        sacc[nf] = __builtin_amdgcn_wmma_f32_16x16x32_f16(
            false, qa[kk], false, bk, (short)0, sacc[nf], false, false);
      }

    float newm[8];
#pragma unroll
    for (int v = 0; v < 8; ++v) newm[v] = mrow[v];
#pragma unroll
    for (int nf = 0; nf < 4; ++nf)
#pragma unroll
      for (int v = 0; v < 8; ++v) {
        float s  = sacc[nf][v] * 0.125f;      // 1/sqrt(64)
        int   qi = qrow0 + g * 8 + v;
        int   ki = j0 + nf * 16 + r;
        if (ki > qi) s = -1e30f;
        sacc[nf][v] = s;
        newm[v] = fmaxf(newm[v], s);
      }
    // half-wave butterfly (xor 1,2,4,8 stays inside the 16-lane C groups)
#pragma unroll
    for (int off = 1; off < 16; off <<= 1)
#pragma unroll
      for (int v = 0; v < 8; ++v)
        newm[v] = fmaxf(newm[v], __shfl_xor(newm[v], off, 32));

    float rsum[8] = {};
#pragma unroll
    for (int nf = 0; nf < 4; ++nf)
#pragma unroll
      for (int v = 0; v < 8; ++v) {
        float p = __expf(sacc[nf][v] - newm[v]);
        sacc[nf][v] = p;
        rsum[v] += p;
      }
#pragma unroll
    for (int off = 1; off < 16; off <<= 1)
#pragma unroll
      for (int v = 0; v < 8; ++v)
        rsum[v] += __shfl_xor(rsum[v], off, 32);

#pragma unroll
    for (int v = 0; v < 8; ++v) {
      float alpha = __expf(mrow[v] - newm[v]);
      lrow[v] = lrow[v] * alpha + rsum[v];
      mrow[v] = newm[v];
#pragma unroll
      for (int nf = 0; nf < 4; ++nf) o[nf][v] *= alpha;
    }

    // -- transpose P through LDS into A-fragment layout --
#pragma unroll
    for (int nf = 0; nf < 4; ++nf)
#pragma unroll
      for (int v = 0; v < 8; ++v)
        Pt[wid][g * 8 + v][nf * 16 + r] = (half_t)sacc[nf][v];
    asm volatile("s_wait_dscnt 0x0" ::: "memory");   // intra-wave LDS RAW

    // -- O += P @ V --
#pragma unroll
    for (int kk = 0; kk < 2; ++kk) {
      union { v16h v; v8h h[2]; } pa;
      pa.h[0] = *(const v8h*)(&Pt[wid][r][kk * 32 + g * 8]);
      pa.h[1] = *(const v8h*)(&Pt[wid][r][kk * 32 + 16 + g * 8]);
#pragma unroll
      for (int nf = 0; nf < 4; ++nf) {
        v16h vb = load_b_pair(&Vt[cur][nf * 16 + r][kk * 32 + g * 8]);
        o[nf] = __builtin_amdgcn_wmma_f32_16x16x32_f16(
            false, pa.v, false, vb, (short)0, o[nf], false, false);
      }
    }
    // no trailing barrier: next iteration's top barrier protects Kt/Vt[cur]
  }

  // -- epilogue: O / l -> Y f16 [B,T,NH,HD] --
  half_t* Yb = Y + (size_t)b * T_SZ * ldq + h * HD;
#pragma unroll
  for (int v = 0; v < 8; ++v) {
    float inv = 1.0f / lrow[v];
    size_t qi = (size_t)(qrow0 + g * 8 + v);
#pragma unroll
    for (int nf = 0; nf < 4; ++nf)
      Yb[qi * ldq + nf * 16 + r] = (half_t)(o[nf][v] * inv);
  }
}

// ---------------------------------------------------------------------------
extern "C" void kernel_launch(void* const* d_in, const int* in_sizes, int n_in,
                              void* d_out, int out_size, void* d_ws, size_t ws_size,
                              hipStream_t stream) {
  (void)in_sizes; (void)n_in; (void)out_size; (void)ws_size;
  const float* x  = (const float*)d_in[0];
  const float* Wq = (const float*)d_in[1];
  const float* Wk = (const float*)d_in[2];
  const float* Wv = (const float*)d_in[3];
  const float* Wo = (const float*)d_in[4];
  float* out = (float*)d_out;

  // workspace layout (halves); total 38 MiB
  half_t* ws = (half_t*)d_ws;
  size_t off = 0;
  half_t* x16  = ws + off; off += (size_t)BT * D_MODEL;
  half_t* wq16 = ws + off; off += (size_t)D_MODEL * D_MODEL;
  half_t* wk16 = ws + off; off += (size_t)(N_KV * HD) * D_MODEL;
  half_t* wv16 = ws + off; off += (size_t)(N_KV * HD) * D_MODEL;
  half_t* wo16 = ws + off; off += (size_t)D_MODEL * D_MODEL;
  half_t* q16  = ws + off; off += (size_t)BT * D_MODEL;
  half_t* k16  = ws + off; off += (size_t)BT * (N_KV * HD);
  half_t* v16  = ws + off; off += (size_t)BT * (N_KV * HD);
  half_t* y16  = ws + off; off += (size_t)BT * D_MODEL;

  const int thr = 256;
  auto cast = [&](const float* s, half_t* d, int n) {
    cast_f32_to_f16<<<(n + thr - 1) / thr, thr, 0, stream>>>(s, d, n);
  };
  cast(x,  x16,  BT * D_MODEL);
  cast(Wq, wq16, D_MODEL * D_MODEL);
  cast(Wk, wk16, N_KV * HD * D_MODEL);
  cast(Wv, wv16, N_KV * HD * D_MODEL);
  cast(Wo, wo16, D_MODEL * D_MODEL);

  // projections: C[M,N] = A[M,K] @ W[N,K]^T
  gemm_nt_wmma<false><<<dim3(BT / 128, D_MODEL / 128), 256, 0, stream>>>(
      x16, wq16, q16, BT, D_MODEL, D_MODEL);
  gemm_nt_wmma<false><<<dim3(BT / 128, (N_KV * HD) / 128), 256, 0, stream>>>(
      x16, wk16, k16, BT, N_KV * HD, D_MODEL);
  gemm_nt_wmma<false><<<dim3(BT / 128, (N_KV * HD) / 128), 256, 0, stream>>>(
      x16, wv16, v16, BT, N_KV * HD, D_MODEL);

  // RoPE
  {
    int nq = BT * N_HEAD * (HD / 2);
    int nk = BT * N_KV * (HD / 2);
    rope_f16<<<(nq + thr - 1) / thr, thr, 0, stream>>>(q16, N_HEAD);
    rope_f16<<<(nk + thr - 1) / thr, thr, 0, stream>>>(k16, N_KV);
  }

  // flash attention
  flash_attn_wmma<<<dim3(T_SZ / 64, N_HEAD, B_SZ), 128, 0, stream>>>(
      q16, k16, v16, y16);

  // output projection (fp32 out)
  gemm_nt_wmma<true><<<dim3(BT / 128, D_MODEL / 128), 256, 0, stream>>>(
      y16, wo16, (void*)out, BT, D_MODEL, D_MODEL);
}
